// NSMModel_86577950752887
// MI455X (gfx1250) — compile-verified, matrix operand order checked by reading, not run on hardware
//
#include <hip/hip_runtime.h>
#include <hip/hip_bf16.h>
#include <math.h>

// ---------------------------------------------------------------------------
// MI455X (gfx1250) implementation.
// All GEMM operands (A and B) are pre-packed into per-lane bf16 WMMA fragment
// layout, so the GEMM inner loop is pure b128 loads + v_wmma_f32_16x16x32_bf16
// (no conversions, no strided access). Packing is fused into passes that
// already touch the data (segment-mean normalize, basis combine).
// ---------------------------------------------------------------------------

#define F_DIM 256
#define HID_DIM 512

typedef __attribute__((ext_vector_type(16))) __bf16 v16bf;
typedef __attribute__((ext_vector_type(8)))  float  v8f;

// Fragment layouts (both A and B): [tile][lane][16 bf16] = 32B per lane.
// A tile = 16 rows x 32 K   : lane L<16 -> row L, K {0..7,16..23}; L+16 -> K {8..15,24..31}
// B tile = 32 K  x 16 cols  : lane L<16 -> col L, same K split.
static __device__ __forceinline__ const v16bf* frag_ptr(const __bf16* P,
                                                        int t0, int t1, int n1,
                                                        int lane) {
  return (const v16bf*)(P + ((((size_t)t0 * n1 + t1) * 32 + lane) << 4));
}

static __device__ __forceinline__ v8f wmma_bf16(v16bf a, v16bf b, v8f c) {
  return __builtin_amdgcn_wmma_f32_16x16x32_bf16(false, a, false, b,
                                                 (short)0, c, false, false);
}

// --- elementwise / utility kernels -----------------------------------------
__global__ __launch_bounds__(256) void fill_f32(float* p, float v, size_t n) {
  size_t i = (size_t)blockIdx.x * 256 + threadIdx.x;
  if (i < n) p[i] = v;
}
__global__ __launch_bounds__(256) void fill_i32(int* p, int v, size_t n) {
  size_t i = (size_t)blockIdx.x * 256 + threadIdx.x;
  if (i < n) p[i] = v;
}

// Pack row-major fp32 B (KxN) into bf16 B-fragment layout [kt][ct][lane][16].
__global__ __launch_bounds__(256) void pack_b(const float* __restrict__ B,
                                              __bf16* __restrict__ out,
                                              int K, int N) {
  size_t i = (size_t)blockIdx.x * 256 + threadIdx.x;
  size_t tot = (size_t)K * N;
  if (i >= tot) return;
  int j    = (int)(i & 15);
  int lane = (int)((i >> 4) & 31);
  size_t t = i >> 9;
  int nct  = N >> 4;
  int ct   = (int)(t % nct);
  int kt   = (int)(t / nct);
  int khi  = (lane >> 4) << 3;
  int koff = (j < 8) ? j : (j + 8);
  int krow = kt * 32 + khi + koff;
  int col  = ct * 16 + (lane & 15);
  out[i] = (__bf16)B[(size_t)krow * N + col];
}

// Pack fp32 A (M rows, lda stride, starting at col_off, K wide) into bf16
// A-fragment layout [mt][kt][lane][16].
__global__ __launch_bounds__(256) void pack_a(const float* __restrict__ A,
                                              int lda, int col_off,
                                              __bf16* __restrict__ out,
                                              int M, int K) {
  size_t i = (size_t)blockIdx.x * 256 + threadIdx.x;
  size_t tot = (size_t)M * K;
  if (i >= tot) return;
  int j    = (int)(i & 15);
  int lane = (int)((i >> 4) & 31);
  size_t t = i >> 9;
  int nkt  = K >> 5;
  int kt   = (int)(t % nkt);
  int mt   = (int)(t / nkt);
  int row  = mt * 16 + (lane & 15);
  int khi  = (lane >> 4) << 3;
  int koff = (j < 8) ? j : (j + 8);
  int col  = kt * 32 + khi + koff;
  out[i] = (__bf16)A[(size_t)row * lda + col_off + col];
}

// Fused: segment-mean normalize agg (n x Reff*F fp32) and emit packed bf16 A.
__global__ __launch_bounds__(256) void rgcn_norm_pack(
    const float* __restrict__ agg, const float* __restrict__ cnt,
    __bf16* __restrict__ out, int Reff, int K /* = Reff*F */, size_t tot) {
  size_t i = (size_t)blockIdx.x * 256 + threadIdx.x;
  if (i >= tot) return;
  int j    = (int)(i & 15);
  int lane = (int)((i >> 4) & 31);
  size_t t = i >> 9;
  int nkt  = K >> 5;
  int kt   = (int)(t % nkt);
  int mt   = (int)(t / nkt);
  int row  = mt * 16 + (lane & 15);
  int khi  = (lane >> 4) << 3;
  int koff = (j < 8) ? j : (j + 8);
  int col  = kt * 32 + khi + koff;
  int r    = col >> 8;                       // relation block within K
  float c  = cnt[(size_t)row * Reff + r];
  float v  = agg[(size_t)row * K + col] / fmaxf(c, 1.0f);
  out[i] = (__bf16)v;
}

// Fused basis combine + pack: pWc[kt][ct][lane][16] <- sum_b comb[r,b]*V[b,f,g]
__global__ __launch_bounds__(256) void build_W_packed(
    const float* __restrict__ V, const float* __restrict__ comb,
    __bf16* __restrict__ out, int Reff) {
  size_t i = (size_t)blockIdx.x * 256 + threadIdx.x;
  size_t tot = (size_t)Reff * F_DIM * F_DIM;
  if (i >= tot) return;
  int j    = (int)(i & 15);
  int lane = (int)((i >> 4) & 31);
  size_t t = i >> 9;
  int nct  = F_DIM >> 4;
  int ct   = (int)(t % nct);
  int kt   = (int)(t / nct);
  int khi  = (lane >> 4) << 3;
  int koff = (j < 8) ? j : (j + 8);
  int krow = kt * 32 + khi + koff;           // r*F + f
  int col  = ct * 16 + (lane & 15);          // g
  int r    = krow >> 8;
  int f    = krow & 255;
  size_t fg = (size_t)f * F_DIM + col;
  float acc = 0.f;
#pragma unroll
  for (int b = 0; b < 5; ++b)
    acc += comb[r * 5 + b] * V[(size_t)b * F_DIM * F_DIM + fg];
  out[i] = (__bf16)acc;
}

// Per-edge confidence-weighted scatter into (dst, rel) segments.
__global__ __launch_bounds__(256) void rgcn_scatter(
    const float* __restrict__ x, const int* __restrict__ src,
    const int* __restrict__ dst, const int* __restrict__ et,
    const float* __restrict__ conf, const float* __restrict__ valid,
    float* __restrict__ agg, float* __restrict__ cnt, int E, int Reff) {
  int e = blockIdx.x;
  if (e >= E) return;
  float v = valid ? valid[e] : 1.0f;
  if (v == 0.0f) return;
  int s = src[e], d = dst[e];
  int r = (et && Reff > 1) ? et[e] : 0;
  float w = conf[e] * v;
  const float* xs = x + (size_t)s * F_DIM;
  float* ap = agg + ((size_t)d * Reff + r) * F_DIM;
  atomicAdd(&ap[threadIdx.x], xs[threadIdx.x] * w);
  if (threadIdx.x == 0) atomicAdd(&cnt[(size_t)d * Reff + r], v);
}

// out(MxN) = A1@B1 + A2@B2 + bias ; all operands packed bf16 fragments.
// Block: 8 waves, each wave computes a 64x16 strip (4 WMMAs share one B frag).
__global__ __launch_bounds__(256) void gemm_dual_bias(
    const __bf16* __restrict__ A1p, int K1len, const __bf16* __restrict__ B1p,
    const __bf16* __restrict__ A2p, int K2len, const __bf16* __restrict__ B2p,
    const float* __restrict__ bias, float* __restrict__ out, int M, int Ncols) {
  const int wave = threadIdx.x >> 5;
  const int lane = threadIdx.x & 31;
  const int mt0  = blockIdx.x * 4;           // 64 rows
  const int col0 = blockIdx.y * 128 + wave * 16;
  if (col0 >= Ncols) return;
  const int ct  = col0 >> 4;
  const int nct = Ncols >> 4;
  v8f c0 = {}, c1 = {}, c2 = {}, c3 = {};
  const int nkt1 = K1len >> 5;
  for (int kt = 0; kt < nkt1; ++kt) {
    v16bf b  = *frag_ptr(B1p, kt, ct, nct, lane);
    v16bf a0 = *frag_ptr(A1p, mt0 + 0, kt, nkt1, lane);
    v16bf a1 = *frag_ptr(A1p, mt0 + 1, kt, nkt1, lane);
    v16bf a2 = *frag_ptr(A1p, mt0 + 2, kt, nkt1, lane);
    v16bf a3 = *frag_ptr(A1p, mt0 + 3, kt, nkt1, lane);
    __builtin_prefetch(frag_ptr(A1p, mt0, kt + 1, nkt1, lane), 0, 0);
    c0 = wmma_bf16(a0, b, c0);
    c1 = wmma_bf16(a1, b, c1);
    c2 = wmma_bf16(a2, b, c2);
    c3 = wmma_bf16(a3, b, c3);
  }
  if (A2p) {
    const int nkt2 = K2len >> 5;
    for (int kt = 0; kt < nkt2; ++kt) {
      v16bf b  = *frag_ptr(B2p, kt, ct, nct, lane);
      v16bf a0 = *frag_ptr(A2p, mt0 + 0, kt, nkt2, lane);
      v16bf a1 = *frag_ptr(A2p, mt0 + 1, kt, nkt2, lane);
      v16bf a2 = *frag_ptr(A2p, mt0 + 2, kt, nkt2, lane);
      v16bf a3 = *frag_ptr(A2p, mt0 + 3, kt, nkt2, lane);
      c0 = wmma_bf16(a0, b, c0);
      c1 = wmma_bf16(a1, b, c1);
      c2 = wmma_bf16(a2, b, c2);
      c3 = wmma_bf16(a3, b, c3);
    }
  }
  const int col   = col0 + (lane & 15);
  const int mbase = mt0 * 16 + ((lane >> 4) << 3);
  const float bv  = bias ? bias[col] : 0.f;
#pragma unroll
  for (int i = 0; i < 8; ++i) {
    out[(size_t)(mbase +  0 + i) * Ncols + col] = c0[i] + bv;
    out[(size_t)(mbase + 16 + i) * Ncols + col] = c1[i] + bv;
    out[(size_t)(mbase + 32 + i) * Ncols + col] = c2[i] + bv;
    out[(size_t)(mbase + 48 + i) * Ncols + col] = c3[i] + bv;
  }
}

// LayerNorm + ReLU, one wave32 per row of 256.
__global__ __launch_bounds__(256) void ln_relu(float* __restrict__ x,
                                               const float* __restrict__ gamma,
                                               const float* __restrict__ beta,
                                               int M) {
  int row  = blockIdx.x * 8 + (threadIdx.x >> 5);
  int lane = threadIdx.x & 31;
  if (row >= M) return;
  float* p = x + (size_t)row * F_DIM;
  float s = 0.f, s2 = 0.f;
  for (int j = lane; j < F_DIM; j += 32) { float v = p[j]; s += v; s2 += v * v; }
#pragma unroll
  for (int o = 16; o; o >>= 1) { s += __shfl_xor(s, o, 32); s2 += __shfl_xor(s2, o, 32); }
  float m   = s * (1.0f / F_DIM);
  float var = s2 * (1.0f / F_DIM) - m * m;
  float inv = rsqrtf(var + 1e-5f);
  for (int j = lane; j < F_DIM; j += 32) {
    float v = (p[j] - m) * inv * gamma[j] + beta[j];
    p[j] = fmaxf(v, 0.f);
  }
}

// One affine-coupling layer (fwd or inverse), 16 rows per block.
// Active half pre-packed (Ap). h1 kept in LDS as packed bf16 fragments so
// stage-2 A fragments are single 32B ds loads.
__global__ __launch_bounds__(256) void coupling_layer(
    float* __restrict__ x, const __bf16* __restrict__ Ap,
    const __bf16* __restrict__ W1p, const float* __restrict__ b1v,
    const __bf16* __restrict__ W2p, const float* __restrict__ b2v,
    int parity, int inversef) {
  __shared__ __align__(32) __bf16 h1p[16][32][16];  // [kt][lane][16] packed
  __shared__ float stb[16][F_DIM];
  const int wave = threadIdx.x >> 5;
  const int lane = threadIdx.x & 31;
  const int row0 = blockIdx.x * 16;
  const int b_off = parity ? 0 : 128;

  // stage 1: h1 = relu(a @ W1 + b1)    (K=128 -> nkt=4, N=512) — 4 tiles/wave
#pragma unroll
  for (int t = 0; t < 4; ++t) {
    const int ct = wave * 4 + t;
    v8f c = {};
#pragma unroll
    for (int kt = 0; kt < 4; ++kt) {
      v16bf a = *frag_ptr(Ap, blockIdx.x, kt, 4, lane);
      v16bf b = *frag_ptr(W1p, kt, ct, HID_DIM >> 4, lane);
      c = wmma_bf16(a, b, c);
    }
    const int col = ct * 16 + (lane & 15);
    const int mb  = (lane >> 4) << 3;
    const float bb = b1v[col];
    const int kt2  = col >> 5;
    const int rem  = col & 31;
    const int l2hi = (rem & 8) ? 16 : 0;
    const int jj   = (rem & 7) + ((rem & 16) ? 8 : 0);
#pragma unroll
    for (int i = 0; i < 8; ++i)
      h1p[kt2][(mb + i) + l2hi][jj] = (__bf16)fmaxf(c[i] + bb, 0.f);
  }
  __syncthreads();
  // stage 2: st = h1 @ W2 + b2        (K=512 -> nkt=16, N=256) — 2 tiles/wave
#pragma unroll
  for (int t = 0; t < 2; ++t) {
    const int ct = wave * 2 + t;
    v8f c = {};
#pragma unroll
    for (int kt = 0; kt < 16; ++kt) {
      v16bf a = *(const v16bf*)&h1p[kt][lane][0];
      v16bf b = *frag_ptr(W2p, kt, ct, F_DIM >> 4, lane);
      c = wmma_bf16(a, b, c);
    }
    const int col = ct * 16 + (lane & 15);
    const int mb  = (lane >> 4) << 3;
    const float bb = b2v[col];
#pragma unroll
    for (int i = 0; i < 8; ++i) stb[mb + i][col] = c[i] + bb;
  }
  __syncthreads();
  // stage 3: update passive half:  fwd: b*exp(s)+t   inv: (b-t)*exp(-s)
#pragma unroll
  for (int q = 0; q < 8; ++q) {
    int e   = threadIdx.x * 8 + q;   // 16 rows x 128 cols
    int row = e >> 7, col = e & 127;
    float s  = stb[row][col];
    float tt = stb[row][col + 128];
    size_t gi = (size_t)(row0 + row) * F_DIM + b_off + col;
    float xb = x[gi];
    x[gi] = inversef ? (xb - tt) * __expf(-s) : xb * __expf(s) + tt;
  }
}

// --- pooling ---------------------------------------------------------------
__global__ __launch_bounds__(256) void pnorm_kernel(const float* __restrict__ p,
                                                    float* __restrict__ out) {
  __shared__ float red[256];
  float v = p[threadIdx.x];
  red[threadIdx.x] = v * v;
  __syncthreads();
  for (int s = 128; s; s >>= 1) {
    if (threadIdx.x < s) red[threadIdx.x] += red[threadIdx.x + s];
    __syncthreads();
  }
  if (threadIdx.x == 0) out[0] = sqrtf(red[0]) + 1e-8f;
}

__global__ __launch_bounds__(256) void score_kernel(const float* __restrict__ x,
                                                    const float* __restrict__ p,
                                                    const float* __restrict__ pn,
                                                    float* __restrict__ key,
                                                    int* __restrict__ idx, int M) {
  int row  = blockIdx.x * 8 + (threadIdx.x >> 5);
  int lane = threadIdx.x & 31;
  if (row >= M) return;
  const float* xr = x + (size_t)row * F_DIM;
  float acc = 0.f;
  for (int j = lane; j < F_DIM; j += 32) acc += xr[j] * p[j];
#pragma unroll
  for (int o = 16; o; o >>= 1) acc += __shfl_xor(acc, o, 32);
  if (lane == 0) { key[row] = acc / pn[0]; idx[row] = row; }
}

// Descending bitonic sort (key,payload) — one compare-exchange stage.
__global__ __launch_bounds__(256) void bitonic_step(float* __restrict__ key,
                                                    int* __restrict__ idx,
                                                    int n, int k, int j) {
  int i = blockIdx.x * 256 + threadIdx.x;
  if (i >= n) return;
  int ixj = i ^ j;
  if (ixj > i) {
    bool desc = ((i & k) == 0);
    float ki = key[i], kj = key[ixj];
    bool sw = desc ? (ki < kj) : (ki > kj);
    if (sw) {
      key[i] = kj; key[ixj] = ki;
      int t = idx[i]; idx[i] = idx[ixj]; idx[ixj] = t;
    }
  }
}

__global__ __launch_bounds__(256) void gate_gather(const float* __restrict__ x,
                                                   const float* __restrict__ key,
                                                   const int* __restrict__ idx,
                                                   float* __restrict__ xp,
                                                   int* __restrict__ perm, int k) {
  int i = blockIdx.x;
  if (i >= k) return;
  int s = idx[i];
  float g = tanhf(key[i]);
  xp[(size_t)i * F_DIM + threadIdx.x] = x[(size_t)s * F_DIM + threadIdx.x] * g;
  if (threadIdx.x == 0) perm[i] = s;
}

__global__ __launch_bounds__(256) void scatter_inv(int* __restrict__ inv,
                                                   const int* __restrict__ perm,
                                                   int k) {
  int i = blockIdx.x * 256 + threadIdx.x;
  if (i < k) inv[perm[i]] = i;
}

__global__ __launch_bounds__(256) void relabel(const int* __restrict__ s,
                                               const int* __restrict__ d,
                                               const float* __restrict__ valid,
                                               const int* __restrict__ inv,
                                               int* __restrict__ s2,
                                               int* __restrict__ d2,
                                               float* __restrict__ v2, int E) {
  int e = blockIdx.x * 256 + threadIdx.x;
  if (e >= E) return;
  int a = inv[s[e]], b = inv[d[e]];
  float v = valid ? valid[e] : 1.0f;
  v2[e] = v * ((a >= 0 && b >= 0) ? 1.0f : 0.0f);
  s2[e] = a > 0 ? a : 0;
  d2[e] = b > 0 ? b : 0;
}

__global__ __launch_bounds__(256) void scatter_rows(float* __restrict__ u,
                                                    const int* __restrict__ perm,
                                                    const float* __restrict__ xs,
                                                    int k) {
  int i = blockIdx.x;
  if (i >= k) return;
  u[(size_t)perm[i] * F_DIM + threadIdx.x] = xs[(size_t)i * F_DIM + threadIdx.x];
}

__global__ __launch_bounds__(256) void colmean(const float* __restrict__ x,
                                               float* __restrict__ out, int M) {
  __shared__ float red[256];
  int f = blockIdx.x;
  float acc = 0.f;
  for (int r = threadIdx.x; r < M; r += 256) acc += x[(size_t)r * F_DIM + f];
  red[threadIdx.x] = acc;
  __syncthreads();
  for (int s = 128; s; s >>= 1) {
    if (threadIdx.x < s) red[threadIdx.x] += red[threadIdx.x + s];
    __syncthreads();
  }
  if (threadIdx.x == 0) out[f] = red[0] / (float)M;
}

__global__ __launch_bounds__(64) void head_kernel(const float* __restrict__ mabs,
                                                  const float* __restrict__ mconc,
                                                  const float* __restrict__ W,
                                                  const float* __restrict__ bb,
                                                  float* __restrict__ out) {
  int c = threadIdx.x;
  if (c >= 40) return;
  float acc = bb[c];
  for (int f = 0; f < F_DIM; ++f)
    acc += 0.5f * (mabs[f] + mconc[f]) * W[f * 40 + c];
  out[c] = acc;
}

// ---------------------------------------------------------------------------
extern "C" void kernel_launch(void* const* d_in, const int* in_sizes, int n_in,
                              void* d_out, int out_size, void* d_ws, size_t ws_size,
                              hipStream_t stream) {
  (void)out_size; (void)ws_size;
  const int N = 16384, E = 262144, F = 256, R = 16;
  const int K1n = N / 2, K2n = N / 4;

  // --- input mapping (JAX pytree alphabetical flatten; hedge on order) -----
  const float* x;  const float* edge_attr;
  const int* edge_index; const int* edge_type;
  int base;
  if (in_sizes[0] == N * F) {            // insertion order: x first
    x          = (const float*)d_in[0];
    edge_index = (const int*)  d_in[1];
    edge_type  = (const int*)  d_in[2];
    edge_attr  = (const float*)d_in[3];
    base = 4;
  } else {                               // alphabetical: edge_attr first, x last
    edge_attr  = (const float*)d_in[0];
    edge_index = (const int*)  d_in[1];
    edge_type  = (const int*)  d_in[2];
    x          = (const float*)d_in[n_in - 1];
    base = 3;
  }
  const int* srcE = edge_index;
  const int* dstE = edge_index + E;

  const float* head_W = (const float*)d_in[base + 0];
  const float* head_b = (const float*)d_in[base + 1];

  struct Cpl  { const float *W1, *W2, *b1, *b2; };
  struct Rg   { const float *V, *Wroot, *beta, *bias, *comb, *gamma; };
  struct Hier { Cpl cf[3]; Cpl ci[3]; const float* pool_p; Rg ra, rb; };

  auto getRg = [&](int off) {
    Rg r;
    r.V     = (const float*)d_in[off + 0];
    r.Wroot = (const float*)d_in[off + 1];
    r.beta  = (const float*)d_in[off + 2];
    r.bias  = (const float*)d_in[off + 3];
    r.comb  = (const float*)d_in[off + 4];
    r.gamma = (const float*)d_in[off + 5];
    return r;
  };
  auto getHier = [&](int off) {
    Hier h;
    for (int i = 0; i < 3; ++i) {
      h.cf[i].W1 = (const float*)d_in[off + i * 4 + 0];
      h.cf[i].W2 = (const float*)d_in[off + i * 4 + 1];
      h.cf[i].b1 = (const float*)d_in[off + i * 4 + 2];
      h.cf[i].b2 = (const float*)d_in[off + i * 4 + 3];
      h.ci[i].W1 = (const float*)d_in[off + 12 + i * 4 + 0];
      h.ci[i].W2 = (const float*)d_in[off + 12 + i * 4 + 1];
      h.ci[i].b1 = (const float*)d_in[off + 12 + i * 4 + 2];
      h.ci[i].b2 = (const float*)d_in[off + 12 + i * 4 + 3];
    }
    h.pool_p = (const float*)d_in[off + 24];
    h.ra = getRg(off + 25);
    h.rb = getRg(off + 31);
    return h;
  };
  Hier L1 = getHier(base + 2);
  Hier L2 = getHier(base + 2 + 37);

  // --- workspace carve -----------------------------------------------------
  char* wp = (char*)d_ws;
  auto carve = [&](size_t bytes) {
    void* p = wp;
    wp += (bytes + 255) & ~(size_t)255;
    return p;
  };
  float*  agg   = (float*)carve((size_t)N * R * F * 4);
  float*  cnt   = (float*)carve((size_t)N * R * 4);
  __bf16* pAgg  = (__bf16*)carve((size_t)N * R * F * 2);
  __bf16* pFeat = (__bf16*)carve((size_t)N * F * 2);
  __bf16* pAct  = (__bf16*)carve((size_t)N * 128 * 2);
  __bf16* pWc   = (__bf16*)carve((size_t)R * F * F * 2);
  __bf16* pWroot= (__bf16*)carve((size_t)F * F * 2);
  __bf16* pW1   = (__bf16*)carve((size_t)12 * 128 * HID_DIM * 2);
  __bf16* pW2   = (__bf16*)carve((size_t)12 * HID_DIM * F * 2);
  float*  featN = (float*)carve((size_t)N * F * 4);
  float*  xp1   = (float*)carve((size_t)K1n * F * 4);
  float*  z1    = (float*)carve((size_t)K1n * F * 4);
  float*  xp2   = (float*)carve((size_t)K2n * F * 4);
  float*  z2    = (float*)carve((size_t)K2n * F * 4);
  float*  key   = (float*)carve((size_t)N * 4);
  int*    idx   = (int*)  carve((size_t)N * 4);
  int*    perm1 = (int*)  carve((size_t)K1n * 4);
  int*    perm2 = (int*)  carve((size_t)K2n * 4);
  int*    inv   = (int*)  carve((size_t)N * 4);
  int*    s1    = (int*)  carve((size_t)E * 4);
  int*    d1    = (int*)  carve((size_t)E * 4);
  float*  v1    = (float*)carve((size_t)E * 4);
  int*    s2    = (int*)  carve((size_t)E * 4);
  int*    d2    = (int*)  carve((size_t)E * 4);
  float*  v2    = (float*)carve((size_t)E * 4);
  float*  u2    = (float*)carve((size_t)K1n * F * 4);
  float*  u1    = (float*)carve((size_t)N * F * 4);
  float*  pn    = (float*)carve(256);
  float*  mabs  = (float*)carve((size_t)F * 4);
  float*  mconc = (float*)carve((size_t)F * 4);

  auto fillf = [&](float* p, float v, size_t n) {
    fill_f32<<<dim3((unsigned)((n + 255) / 256)), dim3(256), 0, stream>>>(p, v, n);
  };
  auto filli = [&](int* p, int v, size_t n) {
    fill_i32<<<dim3((unsigned)((n + 255) / 256)), dim3(256), 0, stream>>>(p, v, n);
  };
  auto packB = [&](const float* B, __bf16* out, int K, int Ncols) {
    size_t tot = (size_t)K * Ncols;
    pack_b<<<dim3((unsigned)((tot + 255) / 256)), dim3(256), 0, stream>>>(B, out, K, Ncols);
  };
  auto packA = [&](const float* A, int lda, int coff, __bf16* out, int M, int K) {
    size_t tot = (size_t)M * K;
    pack_a<<<dim3((unsigned)((tot + 255) / 256)), dim3(256), 0, stream>>>(A, lda, coff, out, M, K);
  };
  auto bitonic = [&](float* k_, int* i_, int n) {
    int blocks = (n + 255) / 256;
    for (int k = 2; k <= n; k <<= 1)
      for (int j = k >> 1; j > 0; j >>= 1)
        bitonic_step<<<dim3(blocks), dim3(256), 0, stream>>>(k_, i_, n, k, j);
  };

  // --- pack all coupling weights once --------------------------------------
  const Hier* Ls[2] = { &L1, &L2 };
  for (int l = 0; l < 2; ++l)
    for (int s = 0; s < 2; ++s)
      for (int i = 0; i < 3; ++i) {
        int slot = l * 6 + s * 3 + i;
        const Cpl& cp = s ? Ls[l]->ci[i] : Ls[l]->cf[i];
        packB(cp.W1, pW1 + (size_t)slot * 128 * HID_DIM, 128, HID_DIM);
        packB(cp.W2, pW2 + (size_t)slot * HID_DIM * F, HID_DIM, F);
      }
  auto cplW1 = [&](int l, int s, int i) { return pW1 + (size_t)(l * 6 + s * 3 + i) * 128 * HID_DIM; };
  auto cplW2 = [&](int l, int s, int i) { return pW2 + (size_t)(l * 6 + s * 3 + i) * HID_DIM * F; };

  // conf-weighted basis R-GCN: scatter-mean + WMMA GEMM + LN + ReLU
  auto run_rgcn = [&](const float* feats, int n, const int* sa, const int* da,
                      const int* et, const float* valid, const Rg& P, int Reff,
                      float* outb) {
    size_t aggTot = (size_t)n * Reff * F;
    fillf(agg, 0.f, aggTot);
    fillf(cnt, 0.f, (size_t)n * Reff);
    build_W_packed<<<dim3((unsigned)(((size_t)Reff * F * F + 255) / 256)),
                     dim3(256), 0, stream>>>(P.V, P.comb, pWc, Reff);
    packB(P.Wroot, pWroot, F, F);
    rgcn_scatter<<<dim3(E), dim3(256), 0, stream>>>(feats, sa, da, et, edge_attr,
                                                    valid, agg, cnt, E, Reff);
    rgcn_norm_pack<<<dim3((unsigned)((aggTot + 255) / 256)), dim3(256), 0,
                    stream>>>(agg, cnt, pAgg, Reff, Reff * F, aggTot);
    packA(feats, F, 0, pFeat, n, F);
    gemm_dual_bias<<<dim3(n / 64, 2), dim3(256), 0, stream>>>(
        pAgg, Reff * F, pWc, pFeat, F, pWroot, P.bias, outb, n, F);
    ln_relu<<<dim3(n / 8), dim3(256), 0, stream>>>(outb, P.gamma, P.beta, n);
  };

  auto run_coupling = [&](float* feats, int n, int l, int setci, bool invf) {
    const Cpl* L = setci ? Ls[l]->ci : Ls[l]->cf;
    for (int step = 0; step < 3; ++step) {
      int i = invf ? 2 - step : step;
      int a_off = (i & 1) ? 128 : 0;
      packA(feats, F, a_off, pAct, n, 128);
      coupling_layer<<<dim3(n / 16), dim3(256), 0, stream>>>(
          feats, pAct, cplW1(l, setci, i), L[i].b1, cplW2(l, setci, i), L[i].b2,
          i & 1, invf ? 1 : 0);
    }
  };

  auto run_pool = [&](const float* feats, int n, int k, const float* pool_p,
                      const int* sa, const int* da, const float* valid,
                      float* xpo, int* permo, int* so, int* dd, float* vo) {
    pnorm_kernel<<<dim3(1), dim3(256), 0, stream>>>(pool_p, pn);
    score_kernel<<<dim3(n / 8), dim3(256), 0, stream>>>(feats, pool_p, pn, key, idx, n);
    bitonic(key, idx, n);
    gate_gather<<<dim3(k), dim3(256), 0, stream>>>(feats, key, idx, xpo, permo, k);
    filli(inv, -1, (size_t)n);
    scatter_inv<<<dim3((k + 255) / 256), dim3(256), 0, stream>>>(inv, permo, k);
    relabel<<<dim3((E + 255) / 256), dim3(256), 0, stream>>>(sa, da, valid, inv,
                                                             so, dd, vo, E);
  };

  // ---- layer 1 "_why" : real edge types (R=16) ----------------------------
  run_rgcn(x, N, srcE, dstE, edge_type, nullptr, L1.ra, R, featN);
  run_coupling(featN, N, 0, 0, false);
  run_pool(featN, N, K1n, L1.pool_p, srcE, dstE, nullptr, xp1, perm1, s1, d1, v1);
  run_rgcn(xp1, K1n, s1, d1, nullptr, v1, L1.rb, 1, z1);    // et0 => Reff=1

  // ---- layer 2 "_why" : edge types all zero (Reff=1 throughout) -----------
  run_rgcn(z1, K1n, s1, d1, nullptr, v1, L2.ra, 1, featN);
  run_coupling(featN, K1n, 1, 0, false);
  run_pool(featN, K1n, K2n, L2.pool_p, s1, d1, v1, xp2, perm2, s2, d2, v2);
  run_rgcn(xp2, K2n, s2, d2, nullptr, v2, L2.rb, 1, z2);

  // ---- abstract feature ---------------------------------------------------
  colmean<<<dim3(F), dim3(256), 0, stream>>>(z2, mabs, K2n);

  // ---- "_what" inverse path ----------------------------------------------
  fillf(u2, 0.f, (size_t)K1n * F);
  scatter_rows<<<dim3(K2n), dim3(256), 0, stream>>>(u2, perm2, z2, K2n);
  run_coupling(u2, K1n, 1, 1, true);

  fillf(u1, 0.f, (size_t)N * F);
  scatter_rows<<<dim3(K1n), dim3(256), 0, stream>>>(u1, perm1, u2, K1n);
  run_coupling(u1, N, 0, 1, true);

  colmean<<<dim3(F), dim3(256), 0, stream>>>(u1, mconc, N);

  // ---- head ---------------------------------------------------------------
  head_kernel<<<dim3(1), dim3(64), 0, stream>>>(mabs, mconc, head_W, head_b,
                                                (float*)d_out);
}
